// SparseLayer_87522843561392
// MI455X (gfx1250) — matrix operand order repaired and checked by reading, not compiled, not run on hardware
//
#include <hip/hip_runtime.h>
#include <stdint.h>

// ---------------------------------------------------------------------------
// out(256 x 8192) = inputs(256 x 8192) @ W_sparse(8192 x 8192, 1M nnz)
// Densify W *transposed* (Wt[m][k], bf16, 128MB, L2-resident) so both GEMM
// operands stage via global_load_async_to_lds_b128 with zero LDS transposes.
// Duplicates handled with global_atomic_pk_add_bf16.
// ---------------------------------------------------------------------------

typedef __attribute__((ext_vector_type(16))) __bf16 v16bf;
typedef __attribute__((ext_vector_type(8)))  __bf16 v8bf;
typedef __attribute__((ext_vector_type(8)))  float  v8f;

#define N_DIM 8192
#define M_DIM 8192
#define BATCH 256
#define NNZ   1048576

static constexpr size_t W_ELEMS = (size_t)N_DIM * (size_t)M_DIM;   // 64M bf16
static constexpr size_t W_BYTES = W_ELEMS * 2;                     // 128 MB
static constexpr size_t A_ELEMS = (size_t)BATCH * (size_t)N_DIM;   // 2M bf16

// LDS tile geometry: K-step 64 (128B of K per row), rows padded to 144B.
// 144 = 16*9 -> b128-aligned; bank stride 36, 36*r mod 64 distinct for r=0..15.
#define ROWB 144
#define ROWE (ROWB / 2)          // 72 elems
#define A_TILE_B (64 * ROWB)     // 9216 B per buffer
#define B_TILE_B (128 * ROWB)    // 18432 B per buffer

__device__ __forceinline__ unsigned short f32_to_bf16_rne(float f) {
  union { float f; unsigned u; } v; v.f = f;
  unsigned u = v.u;
  unsigned r = u + 0x7FFFu + ((u >> 16) & 1u);
  return (unsigned short)(r >> 16);
}

// ---- kernel 1: zero the dense Wt region of the workspace (128 MB) ---------
__global__ __launch_bounds__(256)
void zero_w_kernel(uint4* __restrict__ w) {
  const size_t total = W_BYTES / 16;
  const size_t stride = (size_t)gridDim.x * blockDim.x;
  uint4 z; z.x = 0u; z.y = 0u; z.z = 0u; z.w = 0u;
  for (size_t i = (size_t)blockIdx.x * blockDim.x + threadIdx.x; i < total; i += stride)
    w[i] = z;
}

// ---- kernel 2: convert inputs f32 -> bf16 (row-major, BATCH x N) ----------
__global__ __launch_bounds__(256)
void convert_a_kernel(const float* __restrict__ in, unsigned* __restrict__ a_packed) {
  const size_t i = (size_t)blockIdx.x * blockDim.x + threadIdx.x; // exact: A_ELEMS/2
  const float2 f2 = ((const float2*)in)[i];
  const unsigned lo = f32_to_bf16_rne(f2.x);
  const unsigned hi = f32_to_bf16_rne(f2.y);
  a_packed[i] = lo | (hi << 16);
}

// ---- kernel 3: scatter nnz into dense bf16 Wt[m][k] with pk atomics -------
__global__ __launch_bounds__(256)
void scatter_w_kernel(const long long* __restrict__ idx,
                      const float* __restrict__ wv,
                      unsigned short* __restrict__ Wt) {
  const int k = blockIdx.x * 256 + threadIdx.x;          // grid sized exactly NNZ
  const long long r = idx[2 * k];                        // input feature (k-dim)
  const long long c = idx[2 * k + 1];                    // output column (m-dim)
  const unsigned short b = f32_to_bf16_rne(wv[k]);
  // Wt[c][r]: pk pair along k; other half adds +0.0 (no-op).
  const unsigned data = (r & 1) ? ((unsigned)b << 16) : (unsigned)b;
  const unsigned long long addr =
      (unsigned long long)(size_t)Wt + (((size_t)c * N_DIM + (size_t)(r & ~1LL)) << 1);
  asm volatile("global_atomic_pk_add_bf16 %0, %1, off"
               :: "v"(addr), "v"(data) : "memory");
}

// ---- kernel 4: bf16 WMMA GEMM: out = A(256xN) * Wt^T ----------------------
// 256 threads = 8 waves. Block tile: 64 batch x 128 cols, K-step 64 per buffer.
// Wave w: bi = w&3 (16 batch rows), nj = w>>2 (64-col group -> 4 WMMA tiles).
// Both tiles staged with async global->LDS, ping-pong buffered on ASYNCcnt.

__device__ __forceinline__ v16bf cat16(v8bf lo, v8bf hi) {
  return __builtin_shufflevector(lo, hi, 0,1,2,3,4,5,6,7,8,9,10,11,12,13,14,15);
}

// 8 WMMAs over one staged buffer (two 32-K chunks), fragments loaded
// just-in-time per tile to keep accumulator registers pinned.
__device__ __forceinline__ void compute_tiles(const unsigned short* aptr,
                                              const unsigned short* bptr,
                                              v8f& acc0, v8f& acc1,
                                              v8f& acc2, v8f& acc3) {
#pragma unroll
  for (int kc = 0; kc < 2; ++kc) {
    const int ko = kc * 32;  // elements
    const v8bf alo = *(const v8bf*)(aptr + ko);
    const v8bf ahi = *(const v8bf*)(aptr + ko + 8);
    const v16bf af = cat16(alo, ahi);
    {
      const v16bf bf = cat16(*(const v8bf*)(bptr + ko),
                             *(const v8bf*)(bptr + ko + 8));
      acc0 = __builtin_amdgcn_wmma_f32_16x16x32_bf16(false, af, false, bf,
                                                     (short)0, acc0, false, false);
    }
    {
      const v16bf bf = cat16(*(const v8bf*)(bptr + 16 * ROWE + ko),
                             *(const v8bf*)(bptr + 16 * ROWE + ko + 8));
      acc1 = __builtin_amdgcn_wmma_f32_16x16x32_bf16(false, af, false, bf,
                                                     (short)0, acc1, false, false);
    }
    {
      const v16bf bf = cat16(*(const v8bf*)(bptr + 32 * ROWE + ko),
                             *(const v8bf*)(bptr + 32 * ROWE + ko + 8));
      acc2 = __builtin_amdgcn_wmma_f32_16x16x32_bf16(false, af, false, bf,
                                                     (short)0, acc2, false, false);
    }
    {
      const v16bf bf = cat16(*(const v8bf*)(bptr + 48 * ROWE + ko),
                             *(const v8bf*)(bptr + 48 * ROWE + ko + 8));
      acc3 = __builtin_amdgcn_wmma_f32_16x16x32_bf16(false, af, false, bf,
                                                     (short)0, acc3, false, false);
    }
  }
}

__global__ __launch_bounds__(256)
void spmm_wmma_kernel(const unsigned short* __restrict__ A,
                      const unsigned short* __restrict__ Wt,
                      float* __restrict__ out) {
  __shared__ __align__(16) unsigned short AtileS[2 * A_TILE_B / 2];  // 2 x 64 x 72
  __shared__ __align__(16) unsigned short BtileS[2 * B_TILE_B / 2];  // 2 x 128 x 72

  const int tid  = threadIdx.x;
  const int lane = tid & 31;
  const int wv   = tid >> 5;
  const int bi   = wv & 3;
  const int nj   = wv >> 2;

  const int n0 = blockIdx.x * 128;
  const int b0 = blockIdx.y * 64;

  // --- async staging roles (K-step 64: 128B of K per row) ---
  // A: 64 rows x 128B -> 2 x 16B chunks per thread (64B apart)
  const int ar = tid >> 2, ac = tid & 3;
  const unsigned ldsA0 = (unsigned)(size_t)(&AtileS[0]) + (unsigned)(ar * ROWB + ac * 16);
  const unsigned long long gA0 =
      (unsigned long long)(size_t)A + (((size_t)(b0 + ar) * N_DIM) << 1) + (size_t)(ac * 16);
  // B: 128 rows x 128B -> 4 x 16B chunks per thread (64B contiguous)
  const int br = tid >> 1, bc = (tid & 1) * 64;
  const unsigned ldsB0 = (unsigned)(size_t)(&BtileS[0]) + (unsigned)(br * ROWB + bc);
  const unsigned long long gB0 =
      (unsigned long long)(size_t)Wt + (((size_t)(n0 + br) * N_DIM) << 1) + (size_t)bc;

  // --- fragment base addresses (contiguous K per lane half, same order for A and B) ---
  const int l16 = lane >> 4;
  const int lm  = lane & 15;
  const int klo = l16 * 16;                  // A: K 0..15 / 16..31
  const int kb  = l16 * 16;                  // B: K 0..15 / 16..31
  const unsigned short* aptr = &AtileS[(bi * 16 + lm) * ROWE + klo];
  const unsigned short* bptr = &BtileS[(nj * 64 + lm) * ROWE + kb];

  v8f acc0 = {}, acc1 = {}, acc2 = {}, acc3 = {};

#define ISSUE_TILES(buf, kk)                                                     \
  do {                                                                           \
    const unsigned long long ka = (unsigned long long)((size_t)(kk) << 1);       \
    const unsigned la = ldsA0 + (buf) * A_TILE_B;                                \
    const unsigned lb = ldsB0 + (buf) * B_TILE_B;                                \
    asm volatile("global_load_async_to_lds_b128 %0, %1, off"                     \
                 :: "v"(la), "v"(gA0 + ka) : "memory");                          \
    asm volatile("global_load_async_to_lds_b128 %0, %1, off"                     \
                 :: "v"(la + 64u), "v"(gA0 + ka + 64ull) : "memory");            \
    asm volatile("global_load_async_to_lds_b128 %0, %1, off"                     \
                 :: "v"(lb), "v"(gB0 + ka) : "memory");                          \
    asm volatile("global_load_async_to_lds_b128 %0, %1, off"                     \
                 :: "v"(lb + 16u), "v"(gB0 + ka + 16ull) : "memory");            \
    asm volatile("global_load_async_to_lds_b128 %0, %1, off"                     \
                 :: "v"(lb + 32u), "v"(gB0 + ka + 32ull) : "memory");            \
    asm volatile("global_load_async_to_lds_b128 %0, %1, off"                     \
                 :: "v"(lb + 48u), "v"(gB0 + ka + 48ull) : "memory");            \
  } while (0)

  // ping-pong pipeline, K advances 64 per buffer, explicit 2-buffer unroll
  ISSUE_TILES(0, 0);
  for (int k0 = 0; k0 < N_DIM - 128; k0 += 128) {
    ISSUE_TILES(1, k0 + 64);
    asm volatile("s_wait_asynccnt 0x6" ::: "memory");  // buf0's 6 done (in-order)
    __syncthreads();
    compute_tiles(aptr, bptr, acc0, acc1, acc2, acc3);
    __syncthreads();                                   // reads done before rewrite
    ISSUE_TILES(0, k0 + 128);
    asm volatile("s_wait_asynccnt 0x6" ::: "memory");  // buf1's 6 done
    __syncthreads();
    compute_tiles(aptr + A_TILE_B / 2, bptr + B_TILE_B / 2, acc0, acc1, acc2, acc3);
    __syncthreads();
  }
  // tail: k0 = N_DIM-128 staged in buf0; stage final 64 in buf1
  ISSUE_TILES(1, N_DIM - 64);
  asm volatile("s_wait_asynccnt 0x6" ::: "memory");
  __syncthreads();
  compute_tiles(aptr, bptr, acc0, acc1, acc2, acc3);
  asm volatile("s_wait_asynccnt 0x0" ::: "memory");
  __syncthreads();
  compute_tiles(aptr + A_TILE_B / 2, bptr + B_TILE_B / 2, acc0, acc1, acc2, acc3);

  // C layout: VGPR v -> row m = v + 8*l16, col n = lane&15
  const int cb = b0 + bi * 16 + l16 * 8;
  const int cc = n0 + nj * 64 + lm;
#pragma unroll
  for (int v = 0; v < 8; ++v) {
    float* o = out + (size_t)(cb + v) * M_DIM + cc;
    o[0]  = acc0[v];
    o[16] = acc1[v];
    o[32] = acc2[v];
    o[48] = acc3[v];
  }
}

// ---------------------------------------------------------------------------
extern "C" void kernel_launch(void* const* d_in, const int* in_sizes, int n_in,
                              void* d_out, int out_size, void* d_ws, size_t ws_size,
                              hipStream_t stream) {
  const float*     inputs  = (const float*)d_in[0];
  const long long* indices = (const long long*)d_in[1];   // (NNZ, 2) int64
  const float*     wvals   = (const float*)d_in[2];

  unsigned short* Wt = (unsigned short*)d_ws;                      // 128 MB dense bf16 (M x N)
  unsigned short* A  = (unsigned short*)((char*)d_ws + W_BYTES);   // 4 MB bf16 inputs
  float* out = (float*)d_out;

  zero_w_kernel<<<8192, 256, 0, stream>>>((uint4*)Wt);
  convert_a_kernel<<<(unsigned)(A_ELEMS / 2 / 256), 256, 0, stream>>>(inputs, (unsigned*)A);
  scatter_w_kernel<<<NNZ / 256, 256, 0, stream>>>(indices, wvals, Wt);

  dim3 grid(M_DIM / 128, BATCH / 64);   // 64 x 4 blocks
  spmm_wmma_kernel<<<grid, 256, 0, stream>>>(A, Wt, out);
}
